// GCN_17386027614455
// MI455X (gfx1250) — compile-verified, matrix operand order checked by reading, not run on hardware
//
#include <hip/hip_runtime.h>
#include <cmath>

#define N_NODES 100000
#define N_EDGES 3200000
#define D 256
#define N_CLS 8

typedef __attribute__((ext_vector_type(16))) __bf16 v16bf;
typedef __attribute__((ext_vector_type(8)))  __bf16 v8bf;
typedef __attribute__((ext_vector_type(8)))  float  v8f;
typedef __attribute__((ext_vector_type(4)))  float  f4;

// ---------------- zero fill (vectorized, grid-stride) ----------------
__global__ void zero_f32_kernel(float* __restrict__ p, long n4) {
  long i = (long)blockIdx.x * blockDim.x + threadIdx.x;
  long stride = (long)gridDim.x * blockDim.x;
  f4* p4 = (f4*)p;
  f4 z = {0.f, 0.f, 0.f, 0.f};
  for (; i < n4; i += stride) p4[i] = z;
}

// ---------------- SpMM: out[row] += val * x[col] ----------------
// One wave32 per edge. Lane l gathers floats [4l..4l+3] and [128+4l..128+4l+3]
// of x[col] (2 x b128 loads) and does 8 hardware f32 atomic adds into out[row].
// Both x and out are L2-resident (102.4 MB each < 192 MB L2).
__global__ void spmm_kernel(const int* __restrict__ row, const int* __restrict__ col,
                            const float* __restrict__ val, const float* __restrict__ x,
                            float* __restrict__ out) {
  int gtid  = blockIdx.x * blockDim.x + threadIdx.x;
  int wave  = gtid >> 5;
  int lane  = threadIdx.x & 31;
  int nWaves = (gridDim.x * blockDim.x) >> 5;
  for (int e = wave; e < N_EDGES; e += nWaves) {
    int   r = row[e];
    int   c = col[e];
    float v = val[e];
    const f4* xr = (const f4*)(x + (size_t)c * D);
    float* orow  = out + (size_t)r * D;
    f4 a = xr[lane];
    f4 b = xr[lane + 32];
    int base = lane * 4;
    unsafeAtomicAdd(orow + base + 0, v * a.x);
    unsafeAtomicAdd(orow + base + 1, v * a.y);
    unsafeAtomicAdd(orow + base + 2, v * a.z);
    unsafeAtomicAdd(orow + base + 3, v * a.w);
    unsafeAtomicAdd(orow + 128 + base + 0, v * b.x);
    unsafeAtomicAdd(orow + 128 + base + 1, v * b.y);
    unsafeAtomicAdd(orow + 128 + base + 2, v * b.z);
    unsafeAtomicAdd(orow + 128 + base + 3, v * b.w);
  }
}

// ---------------- W1 -> bf16 transpose (W1T[n][k] = W1[k][n]) ----------------
__global__ void prep_w1t_kernel(const float* __restrict__ W1, __bf16* __restrict__ W1T) {
  int idx = blockIdx.x * blockDim.x + threadIdx.x;  // 65536 threads
  int n = idx & (D - 1);
  int k = idx >> 8;
  W1T[(size_t)n * D + k] = (__bf16)W1[(size_t)k * D + n];
}

// ---------------- H2 = relu(H1 @ W1 + b1) via v_wmma_f32_16x16x32_bf16 ----------------
// One wave per 16x16 output tile. 100000 waves total (6250 M-tiles x 16 N-tiles).
// A fragment (ISA 7.12.2, 16-bit A 16x32): lane L -> row M = L%16,
//   element i -> K = kb + (i<8 ? 0 : 16) + (L/16)*8 + (i%8).
// B fragment mirrors A with column N in place of M, sourced from bf16 W1T.
// C/D layout: VGPR r, lanes 0-15 -> (M=r, N=lane); lanes 16-31 -> (M=8+r, N=lane-16).
__global__ void gemm1_wmma_kernel(const float* __restrict__ H1, const __bf16* __restrict__ W1T,
                                  const float* __restrict__ b1, float* __restrict__ H2) {
  int gtid  = blockIdx.x * blockDim.x + threadIdx.x;
  int wave  = gtid >> 5;            // 0..99999
  int lane  = threadIdx.x & 31;
  int tileM = wave >> 4;            // 0..6249
  int tileN = wave & 15;            // 0..15
  int half  = lane >> 4;
  int l15   = lane & 15;
  int arow  = tileM * 16 + l15;
  int ncol  = tileN * 16 + l15;
  const float*  ap = H1  + (size_t)arow * D;
  const __bf16* bp = W1T + (size_t)ncol * D;

  v8f acc = {};
#pragma unroll
  for (int kb = 0; kb < D; kb += 32) {
    int ka = kb + half * 8;
    f4 a0 = *(const f4*)(ap + ka);
    f4 a1 = *(const f4*)(ap + ka + 4);
    f4 a2 = *(const f4*)(ap + ka + 16);
    f4 a3 = *(const f4*)(ap + ka + 20);
    v16bf A;
    A[0]  = (__bf16)a0.x; A[1]  = (__bf16)a0.y; A[2]  = (__bf16)a0.z; A[3]  = (__bf16)a0.w;
    A[4]  = (__bf16)a1.x; A[5]  = (__bf16)a1.y; A[6]  = (__bf16)a1.z; A[7]  = (__bf16)a1.w;
    A[8]  = (__bf16)a2.x; A[9]  = (__bf16)a2.y; A[10] = (__bf16)a2.z; A[11] = (__bf16)a2.w;
    A[12] = (__bf16)a3.x; A[13] = (__bf16)a3.y; A[14] = (__bf16)a3.z; A[15] = (__bf16)a3.w;

    v8bf blo = *(const v8bf*)(bp + ka);        // 16B aligned
    v8bf bhi = *(const v8bf*)(bp + ka + 16);
    v16bf B;
#pragma unroll
    for (int i = 0; i < 8; i++) { B[i] = blo[i]; B[i + 8] = bhi[i]; }

    acc = __builtin_amdgcn_wmma_f32_16x16x32_bf16(
        /*neg_a=*/false, A, /*neg_b=*/false, B,
        /*c_mod=*/(short)0, acc, /*reuse_a=*/false, /*reuse_b=*/false);
  }

  float bias = b1[ncol];
  float* op = H2 + ((size_t)(tileM * 16 + half * 8)) * D + (size_t)tileN * 16 + l15;
#pragma unroll
  for (int r = 0; r < 8; r++) {
    float o = acc[r] + bias;
    o = o > 0.f ? o : 0.f;                     // ReLU
    op[(size_t)r * D] = o;
  }
}

// ---------------- head: logits = H3 @ W2 + b2, then log_softmax ----------------
// One thread per node; W2/b2 indices are wave-uniform -> scalar (SMEM) loads.
__global__ void head_kernel(const float* __restrict__ H3, const float* __restrict__ W2,
                            const float* __restrict__ b2, float* __restrict__ out) {
  int node = blockIdx.x * blockDim.x + threadIdx.x;
  if (node >= N_NODES) return;
  const float* h = H3 + (size_t)node * D;
  float acc[N_CLS];
#pragma unroll
  for (int c = 0; c < N_CLS; c++) acc[c] = b2[c];
  for (int k = 0; k < D; k += 4) {
    f4 hv = *(const f4*)(h + k);
#pragma unroll
    for (int c = 0; c < N_CLS; c++) {
      acc[c] += hv.x * W2[(k + 0) * N_CLS + c];
      acc[c] += hv.y * W2[(k + 1) * N_CLS + c];
      acc[c] += hv.z * W2[(k + 2) * N_CLS + c];
      acc[c] += hv.w * W2[(k + 3) * N_CLS + c];
    }
  }
  float m = acc[0];
#pragma unroll
  for (int c = 1; c < N_CLS; c++) m = fmaxf(m, acc[c]);
  float s = 0.f;
#pragma unroll
  for (int c = 0; c < N_CLS; c++) s = s + __expf(acc[c] - m);
  float lse = __logf(s);
  float* orow = out + (size_t)node * N_CLS;
#pragma unroll
  for (int c = 0; c < N_CLS; c++) orow[c] = acc[c] - m - lse;
}

// ---------------- driver ----------------
extern "C" void kernel_launch(void* const* d_in, const int* in_sizes, int n_in,
                              void* d_out, int out_size, void* d_ws, size_t ws_size,
                              hipStream_t stream) {
  const float* x       = (const float*)d_in[0];
  const int*   adj_row = (const int*)  d_in[1];
  const int*   adj_col = (const int*)  d_in[2];
  const float* adj_val = (const float*)d_in[3];
  const float* W1      = (const float*)d_in[4];
  const float* b1      = (const float*)d_in[5];
  const float* W2      = (const float*)d_in[6];
  const float* b2      = (const float*)d_in[7];
  float* out = (float*)d_out;

  const size_t nodeF = (size_t)N_NODES * D;          // 25.6M floats
  float*  H1  = (float*)d_ws;                        // [N,D] f32 (reused as H3)
  float*  H2  = H1 + nodeF;                          // [N,D] f32
  __bf16* W1T = (__bf16*)(H2 + nodeF);               // [D,D] bf16

  const long n4 = (long)(nodeF / 4);

  // 1) H1 = 0 ; H1 += spmm(adj, x)
  zero_f32_kernel<<<4096, 256, 0, stream>>>(H1, n4);
  spmm_kernel<<<65536, 256, 0, stream>>>(adj_row, adj_col, adj_val, x, H1);

  // 2) W1T = bf16(W1^T)
  prep_w1t_kernel<<<D * D / 256, 256, 0, stream>>>(W1, W1T);

  // 3) H2 = relu(H1 @ W1 + b1)   (WMMA bf16 -> f32)
  gemm1_wmma_kernel<<<12500, 256, 0, stream>>>(H1, W1T, b1, H2);

  // 4) H3 (= H1 region) = 0 ; H3 += spmm(adj, H2)
  zero_f32_kernel<<<4096, 256, 0, stream>>>(H1, n4);
  spmm_kernel<<<65536, 256, 0, stream>>>(adj_row, adj_col, adj_val, H2, H1);

  // 5) out = log_softmax(H3 @ W2 + b2)
  head_kernel<<<(N_NODES + 255) / 256, 256, 0, stream>>>(H1, W2, b2, out);
}
